// Block_6476810682965
// MI455X (gfx1250) — matrix operand order repaired and checked by reading, not compiled, not run on hardware
//
#include <hip/hip_runtime.h>
#include <hip/hip_bf16.h>

// ---------------------------------------------------------------------------
// Transformer block for MI455X (gfx1250, wave32, WMMA).
// All matmuls run through v_wmma_f32_16x16x32_f16 with f32 accumulation.
// GEMM: 128x128 block tile, 8 waves (2/SIMD32), 32x64 per wave, LDS-staged
// with (if available) GLOBAL_LOAD_ASYNC_TO_LDS_B128 double buffering.
// Attention is a flash-style online-softmax kernel.
// ---------------------------------------------------------------------------

#define B_  512
#define T_  256
#define C_  512
#define H_  4
#define HS_ 128
#define FF_ 2048
#define NF_ 2
#define BT_ (B_*T_)   // 131072 token rows

typedef _Float16 h16;
typedef __attribute__((ext_vector_type(16))) _Float16 v16h;
typedef __attribute__((ext_vector_type(8)))  _Float16 v8h;
typedef __attribute__((ext_vector_type(8)))  float    v8f;
typedef int v4i __attribute__((vector_size(4 * sizeof(int))));  // builtin pointee type

#if __has_builtin(__builtin_amdgcn_global_load_async_to_lds_b128) && \
    __has_builtin(__builtin_amdgcn_s_wait_asynccnt)
#define ASYNC_LDS 1
#endif

// Generic pointers with the builtin's exact pointee type; clang's Sema relaxes
// the builtin's address spaces (global/LDS) and inserts the addrspacecasts.
#define GPTR(p) ((v4i*)(void*)(p))
#define LPTR(p) ((v4i*)(p))

static __device__ __forceinline__ v16h cat8(v8h lo, v8h hi) {
  return __builtin_shufflevector(lo, hi, 0,1,2,3,4,5,6,7,8,9,10,11,12,13,14,15);
}

// ---------------------------------------------------------------------------
// Weight transpose + fp32->fp16 convert:  dst[b][n*ldd + k] = scale*src[b][k*N+n]
// ---------------------------------------------------------------------------
__global__ void transpose_cvt_kernel(const float* __restrict__ src, h16* __restrict__ dst,
                                     int K, int N, long srcBatch, long dstBatch,
                                     int ldd, float scale) {
  __shared__ float tile[32][33];
  src += (size_t)blockIdx.z * srcBatch;
  dst += (size_t)blockIdx.z * dstBatch;
  const int n0 = blockIdx.x * 32, k0 = blockIdx.y * 32;
#pragma unroll
  for (int i = 0; i < 4; i++) {
    int k = k0 + threadIdx.y + i * 8;
    int n = n0 + threadIdx.x;
    tile[threadIdx.y + i * 8][threadIdx.x] = src[(size_t)k * N + n];
  }
  __syncthreads();
#pragma unroll
  for (int i = 0; i < 4; i++) {
    int n = n0 + threadIdx.y + i * 8;
    int k = k0 + threadIdx.x;
    dst[(size_t)n * ldd + k] = (h16)(scale * tile[threadIdx.x][threadIdx.y + i * 8]);
  }
}

// ---------------------------------------------------------------------------
// RMSNorm (scalar weight): one 256-thread block per row of C_=512.
// Writes f16 copy (GEMM A operand) and optionally f32 (may alias X: in-place).
// ---------------------------------------------------------------------------
__global__ __launch_bounds__(256)
void rmsnorm_kernel(const float* __restrict__ X, const float* __restrict__ w,
                    h16* __restrict__ Yh, float* __restrict__ Yf) {
  __shared__ float partial[8];
  const size_t row = blockIdx.x;
  const int tid = threadIdx.x;
  float v0 = X[row * C_ + tid];
  float v1 = X[row * C_ + tid + 256];
  float ss = v0 * v0 + v1 * v1;
#pragma unroll
  for (int off = 16; off >= 1; off >>= 1) ss += __shfl_xor(ss, off, 32);
  if ((tid & 31) == 0) partial[tid >> 5] = ss;
  __syncthreads();
  float tot = 0.f;
#pragma unroll
  for (int i = 0; i < 8; i++) tot += partial[i];
  const float r = w[0] * rsqrtf(tot * (1.0f / C_) + 1e-8f);
  Yh[row * C_ + tid]       = (h16)(v0 * r);
  Yh[row * C_ + tid + 256] = (h16)(v1 * r);
  if (Yf) {
    Yf[row * C_ + tid]       = v0 * r;
    Yf[row * C_ + tid + 256] = v1 * r;
  }
}

// ---------------------------------------------------------------------------
// Generic WMMA GEMM:  D[M,N] = A[M,K](f16,row) * Bt[N,K](f16)  + epilogue
// 256 threads = 8 waves. Block tile 128x128, BK=32. Wave (wm,wn) owns a
// 32x64 sub-tile: 2 A-frags x 4 B-frags -> 8 WMMAs per K-step.
// ---------------------------------------------------------------------------
#define EPI_F16          0
#define EPI_BIAS_F16     1
#define EPI_BIAS_RES_F32 2
#define EPI_GELU_F16     3
#define EPI_RES_F32      4

template <int EPI>
__global__ __launch_bounds__(256)
void gemm_wmma_kernel(const h16* __restrict__ A,  long strideA, int lda,
                      const h16* __restrict__ Bt, long strideB, int ldb,
                      h16* __restrict__ Dh, float* __restrict__ Df, long strideD, int ldd,
                      const float* __restrict__ bias, long strideBias,
                      const float* __restrict__ Res, int ldres,
                      int M, int N, int K) {
#ifdef ASYNC_LDS
  __shared__ h16 As[2][128 * 32];   // double-buffered: async DMA fills buf^1
  __shared__ h16 Bs[2][128 * 32];   // while WMMAs consume buf
#else
  __shared__ h16 As[1][128 * 32];
  __shared__ h16 Bs[1][128 * 32];
#endif
  const int tid  = threadIdx.x;
  const int wave = tid >> 5;
  const int lane = tid & 31;
  const int g    = lane >> 4;
  const int nn   = lane & 15;
  const int wm   = wave & 3;    // M quadrant (32 rows)
  const int wn   = wave >> 2;   // N half (64 cols)
  const int z    = blockIdx.z;
  A  += (size_t)z * strideA;
  Bt += (size_t)z * strideB;
  const size_t dOff = (size_t)z * strideD;
  const int n0 = blockIdx.x * 128;
  const int m0 = blockIdx.y * 128;

  const v8f zero8 = {0.f, 0.f, 0.f, 0.f, 0.f, 0.f, 0.f, 0.f};
  v8f acc[2][4];
#pragma unroll
  for (int mt = 0; mt < 2; mt++)
#pragma unroll
    for (int t = 0; t < 4; t++) acc[mt][t] = zero8;

  // 128x32 tile = 512 b128 chunks per matrix; each thread owns 2 (row=c>>2, blk=c&3)
  auto compute = [&](const h16* __restrict__ as, const h16* __restrict__ bs) {
    v16h af[2], bf[4];
#pragma unroll
    for (int mt = 0; mt < 2; mt++) {
      const int row = wm * 32 + mt * 16 + nn;  // A: lane holds row M = lane&15
      af[mt] = cat8(*(const v8h*)&as[row * 32 + 8 * g],
                    *(const v8h*)&as[row * 32 + 16 + 8 * g]);
    }
#pragma unroll
    for (int t = 0; t < 4; t++) {
      const int row = wn * 64 + t * 16 + nn;   // B: lane holds col N, K=8g..8g+7, 16+8g..16+8g+7
      bf[t] = cat8(*(const v8h*)&bs[row * 32 + 8 * g],
                   *(const v8h*)&bs[row * 32 + 16 + 8 * g]);
    }
#pragma unroll
    for (int mt = 0; mt < 2; mt++)
#pragma unroll
      for (int t = 0; t < 4; t++)
        acc[mt][t] = __builtin_amdgcn_wmma_f32_16x16x32_f16(
            false, af[mt], false, bf[t], (short)0, acc[mt][t], false, false);
  };

#ifdef ASYNC_LDS
  auto issueTile = [&](int k0, int buf) {
#pragma unroll
    for (int i = 0; i < 2; i++) {
      int c = tid + i * 256;
      int row = c >> 2, blk = c & 3;
      __builtin_amdgcn_global_load_async_to_lds_b128(
          GPTR(&A[(size_t)(m0 + row) * lda + k0 + blk * 8]),
          LPTR(&As[buf][row * 32 + blk * 8]), 0, 0);
      __builtin_amdgcn_global_load_async_to_lds_b128(
          GPTR(&Bt[(size_t)(n0 + row) * ldb + k0 + blk * 8]),
          LPTR(&Bs[buf][row * 32 + blk * 8]), 0, 0);
    }
  };
  issueTile(0, 0);
  int ib = 0;
  for (int k0 = 0; k0 < K; k0 += 32, ib ^= 1) {
    if (k0 + 32 < K) {
      issueTile(k0 + 32, ib ^ 1);                 // prefetch next tile into buf^1
      __builtin_amdgcn_s_wait_asynccnt(4);        // wait only this tile's 4 DMAs
    } else {
      __builtin_amdgcn_s_wait_asynccnt(0);
    }
    __syncthreads();                              // all waves' DMAs landed
    compute(As[ib], Bs[ib]);
    __syncthreads();                              // readers done before reuse
  }
#else
  uint4 ra[2], rb[2];
  auto loadRegs = [&](int k0) {
#pragma unroll
    for (int i = 0; i < 2; i++) {
      int c = tid + i * 256;
      int row = c >> 2, blk = c & 3;
      ra[i] = *(const uint4*)&A[(size_t)(m0 + row) * lda + k0 + blk * 8];
      rb[i] = *(const uint4*)&Bt[(size_t)(n0 + row) * ldb + k0 + blk * 8];
    }
  };
  loadRegs(0);
  for (int k0 = 0; k0 < K; k0 += 32) {
    __syncthreads();
#pragma unroll
    for (int i = 0; i < 2; i++) {
      int c = tid + i * 256;
      int row = c >> 2, blk = c & 3;
      *(uint4*)&As[0][row * 32 + blk * 8] = ra[i];
      *(uint4*)&Bs[0][row * 32 + blk * 8] = rb[i];
    }
    __syncthreads();
    if (k0 + 32 < K) loadRegs(k0 + 32);           // global latency under WMMAs
    if (k0 + 64 < K) {
      __builtin_prefetch(&A[(size_t)(m0 + (tid & 127)) * lda + k0 + 64], 0, 1);
      __builtin_prefetch(&Bt[(size_t)(n0 + (tid & 127)) * ldb + k0 + 64], 0, 1);
    }
    compute(As[0], Bs[0]);
  }
#endif

  // epilogue: C/D layout — VGPR r holds row M = 8g + r, col N = lane&15
#pragma unroll
  for (int mt = 0; mt < 2; mt++) {
#pragma unroll
    for (int t = 0; t < 4; t++) {
      const int gn = n0 + wn * 64 + t * 16 + nn;
#pragma unroll
      for (int r = 0; r < 8; r++) {
        const int gm = m0 + wm * 32 + mt * 16 + 8 * g + r;
        float val = acc[mt][t][r];
        if (EPI == EPI_BIAS_F16 || EPI == EPI_BIAS_RES_F32)
          val += bias[(size_t)z * strideBias + gn];
        if (EPI == EPI_BIAS_RES_F32 || EPI == EPI_RES_F32)
          val += Res[(size_t)gm * ldres + gn];
        if (EPI == EPI_GELU_F16)
          val = 0.5f * val * (1.0f + erff(val * 0.70710678118654752f));
        if (EPI == EPI_F16 || EPI == EPI_BIAS_F16 || EPI == EPI_GELU_F16)
          Dh[dOff + (size_t)gm * ldd + gn] = (h16)val;
        else
          Df[dOff + (size_t)gm * ldd + gn] = val;
      }
    }
  }
}

// ---------------------------------------------------------------------------
// Flash attention: grid (T/64, B*H), 128 threads (4 waves, 16 q-rows each).
// Q fragments live in registers; 64-row K / Vt chunks staged in LDS (40KB).
// Online softmax; causal mask applied only on the diagonal chunk.
// q,k,v layout: [H][B*T][HS] f16. Output: [B*T][C] f16 (head-concat layout).
// ---------------------------------------------------------------------------
__global__ __launch_bounds__(128)
void attention_kernel(const h16* __restrict__ Q, const h16* __restrict__ K,
                      const h16* __restrict__ V, h16* __restrict__ O, float scale) {
  __shared__ h16 Ks[64 * 128];   // K chunk, [s][d]
  __shared__ h16 Vt[128 * 64];   // V chunk transposed, [d][s]
  __shared__ h16 Ps[4][16 * 64]; // per-wave P tile
  const int tid  = threadIdx.x;
  const int wave = tid >> 5, lane = tid & 31, g = lane >> 4, nn = lane & 15;
  const int qb = blockIdx.x;
  const int bh = blockIdx.y;
  const int b  = bh >> 2;   // H_ == 4
  const int h  = bh & 3;
  const size_t headOff = (size_t)h * BT_ * HS_;
  const size_t rowBase = (size_t)b * T_;
  const h16* Qp = Q + headOff;
  const h16* Kp = K + headOff;
  const h16* Vp = V + headOff;

  // resident Q fragments: 4 K-steps across HS=128
  const int qrow = qb * 64 + wave * 16 + nn;   // A-frag row M = lane&15
  v16h aq[4];
#pragma unroll
  for (int kk = 0; kk < 4; kk++) {
    const h16* qptr = Qp + (rowBase + qrow) * HS_ + kk * 32;
    aq[kk] = cat8(*(const v8h*)(qptr + 8 * g), *(const v8h*)(qptr + 16 + 8 * g));
  }

  const v8f zero8 = {0.f, 0.f, 0.f, 0.f, 0.f, 0.f, 0.f, 0.f};
  v8f o[8];
#pragma unroll
  for (int t = 0; t < 8; t++) o[t] = zero8;
  float mrun[8], lrun[8];
#pragma unroll
  for (int r = 0; r < 8; r++) { mrun[r] = -1e30f; lrun[r] = 0.f; }

  const int nchunks = qb + 1;   // causal: only chunks at/below the diagonal
  for (int cc = 0; cc < nchunks; ++cc) {
    const int c0 = cc * 64;
    __syncthreads();
    // stage K chunk (copy) and V chunk (transpose) into LDS
#pragma unroll
    for (int i = 0; i < 8; i++) {
      int c = tid + i * 128;
      int row = c >> 4, blk = c & 15;
      *(uint4*)&Ks[row * 128 + blk * 8] =
          *(const uint4*)&Kp[(rowBase + c0 + row) * HS_ + blk * 8];
      uint4 vv = *(const uint4*)&Vp[(rowBase + c0 + row) * HS_ + blk * 8];
      const h16* vp = (const h16*)&vv;
#pragma unroll
      for (int j = 0; j < 8; j++) Vt[(blk * 8 + j) * 64 + row] = vp[j];
    }
    __syncthreads();

    // scores: S = Q K^T (Bt form = K row-major), 4 s-tiles x 4 K-steps
    v8f s[4];
#pragma unroll
    for (int t = 0; t < 4; t++) s[t] = zero8;
#pragma unroll
    for (int kk = 0; kk < 4; kk++) {
#pragma unroll
      for (int t = 0; t < 4; t++) {
        const h16* bp = &Ks[(t * 16 + nn) * 128 + kk * 32 + 8 * g];
        v16h bf = cat8(*(const v8h*)bp, *(const v8h*)(bp + 16));
        s[t] = __builtin_amdgcn_wmma_f32_16x16x32_f16(
            false, aq[kk], false, bf, (short)0, s[t], false, false);
      }
    }
    const bool diag = (c0 == qb * 64);
#pragma unroll
    for (int t = 0; t < 4; t++) {
#pragma unroll
      for (int r = 0; r < 8; r++) {
        float sv = s[t][r] * scale;
        if (diag) {
          int trow = qb * 64 + wave * 16 + 8 * g + r;
          int scol = c0 + t * 16 + nn;
          if (scol > trow) sv = -1e30f;
        }
        s[t][r] = sv;
      }
    }
    // online softmax (row stats reduced across the 16 lanes holding that row)
    float pm[8];
#pragma unroll
    for (int r = 0; r < 8; r++) {
      float mx = fmaxf(fmaxf(s[0][r], s[1][r]), fmaxf(s[2][r], s[3][r]));
      mx = fmaxf(mx, __shfl_xor(mx, 1, 32));
      mx = fmaxf(mx, __shfl_xor(mx, 2, 32));
      mx = fmaxf(mx, __shfl_xor(mx, 4, 32));
      mx = fmaxf(mx, __shfl_xor(mx, 8, 32));
      const float newm = fmaxf(mrun[r], mx);
      const float corr = __expf(mrun[r] - newm);
      mrun[r] = newm;
      lrun[r] *= corr;
#pragma unroll
      for (int t = 0; t < 8; t++) o[t][r] *= corr;
      pm[r] = newm;
    }
    // P = exp(S - m); spill f16 P tile per wave, accumulate row sums
#pragma unroll
    for (int r = 0; r < 8; r++) {
      float rs = 0.f;
#pragma unroll
      for (int t = 0; t < 4; t++) {
        float p = __expf(s[t][r] - pm[r]);
        rs += p;
        Ps[wave][(8 * g + r) * 64 + t * 16 + nn] = (h16)p;
      }
      rs += __shfl_xor(rs, 1, 32);
      rs += __shfl_xor(rs, 2, 32);
      rs += __shfl_xor(rs, 4, 32);
      rs += __shfl_xor(rs, 8, 32);
      lrun[r] += rs;
    }
    __syncthreads();
    // O += P * V : K-dim = 64 (2 steps), 8 d-tiles; Bt form = Vt[d][s]
#pragma unroll
    for (int ks = 0; ks < 2; ks++) {
      const h16* pr = &Ps[wave][nn * 64 + ks * 32];
      v16h ap = cat8(*(const v8h*)(pr + 8 * g), *(const v8h*)(pr + 16 + 8 * g));
#pragma unroll
      for (int t = 0; t < 8; t++) {
        const h16* vptr = &Vt[(t * 16 + nn) * 64 + ks * 32 + 8 * g];
        v16h bf = cat8(*(const v8h*)vptr, *(const v8h*)(vptr + 16));
        o[t] = __builtin_amdgcn_wmma_f32_16x16x32_f16(
            false, ap, false, bf, (short)0, o[t], false, false);
      }
    }
  }

  // normalize + write head-concat output [B*T][C] f16
#pragma unroll
  for (int r = 0; r < 8; r++) {
    const float inv = 1.0f / lrun[r];
    const size_t grow = rowBase + qb * 64 + wave * 16 + 8 * g + r;
#pragma unroll
    for (int t = 0; t < 8; t++)
      O[grow * C_ + h * HS_ + t * 16 + nn] = (h16)(o[t][r] * inv);
  }
}

// ---------------------------------------------------------------------------
// Host-side orchestration
// ---------------------------------------------------------------------------
extern "C" void kernel_launch(void* const* d_in, const int* in_sizes, int n_in,
                              void* d_out, int out_size, void* d_ws, size_t ws_size,
                              hipStream_t stream) {
  const float* x     = (const float*)d_in[0];
  const float* Wq    = (const float*)d_in[1];
  const float* Wk    = (const float*)d_in[2];
  const float* bk    = (const float*)d_in[3];
  const float* Wv    = (const float*)d_in[4];
  const float* Wproj = (const float*)d_in[5];
  const float* bproj = (const float*)d_in[6];
  const float* w1    = (const float*)d_in[7];
  const float* w2    = (const float*)d_in[8];
  const float* Wi    = (const float*)d_in[9];
  const float* Wo    = (const float*)d_in[10];
  float* out = (float*)d_out;
  h16*   ws  = (h16*)d_ws;

  // workspace layout (h16 element offsets); overlapped lifetimes:
  //   buf1: h, then attn-out.  q-region reused for x2(f16).  k+v reused for hmid.
  const size_t o_wq   = 0;
  const size_t o_wk   = o_wq  + (size_t)H_ * HS_ * C_;
  const size_t o_wv   = o_wk  + (size_t)H_ * HS_ * C_;
  const size_t o_wp   = o_wv  + (size_t)H_ * HS_ * C_;
  const size_t o_wi   = o_wp  + (size_t)C_ * C_;
  const size_t o_wo   = o_wi  + (size_t)NF_ * FF_ * C_;
  const size_t o_buf1 = o_wo  + (size_t)C_ * NF_ * FF_;
  const size_t o_q    = o_buf1 + (size_t)BT_ * C_;
  const size_t o_k    = o_q   + (size_t)H_ * BT_ * HS_;
  const size_t o_v    = o_k   + (size_t)H_ * BT_ * HS_;
  (void)in_sizes; (void)n_in; (void)out_size; (void)ws_size;

  const dim3 tb(32, 8);
  // --- weight convert+transpose to f16 [N][K] ---
  transpose_cvt_kernel<<<dim3(HS_/32, C_/32, H_), tb, 0, stream>>>(
      Wq, ws + o_wq, C_, HS_, (long)C_*HS_, (long)HS_*C_, C_, 1.0f);
  transpose_cvt_kernel<<<dim3(HS_/32, C_/32, H_), tb, 0, stream>>>(
      Wk, ws + o_wk, C_, HS_, (long)C_*HS_, (long)HS_*C_, C_, 1.0f);
  transpose_cvt_kernel<<<dim3(HS_/32, C_/32, H_), tb, 0, stream>>>(
      Wv, ws + o_wv, C_, HS_, (long)C_*HS_, (long)HS_*C_, C_, 1.0f);
  transpose_cvt_kernel<<<dim3(C_/32, C_/32, 1), tb, 0, stream>>>(
      Wproj, ws + o_wp, C_, C_, 0L, 0L, C_, 1.0f);
  transpose_cvt_kernel<<<dim3(FF_/32, C_/32, NF_), tb, 0, stream>>>(
      Wi, ws + o_wi, C_, FF_, (long)C_*FF_, (long)FF_*C_, C_, 1.0f);
  // Wo packed as [C][NF*FF] with the branch-mean 0.5 folded in
  transpose_cvt_kernel<<<dim3(C_/32, FF_/32, NF_), tb, 0, stream>>>(
      Wo, ws + o_wo, FF_, C_, (long)FF_*C_, (long)FF_, NF_*FF_, 0.5f);

  // --- h = rmsnorm(x, w1) -> f16 ---
  rmsnorm_kernel<<<BT_, 256, 0, stream>>>(x, w1, ws + o_buf1, nullptr);

  // --- q,k,v per-head GEMMs: [BT,C] x [HS,C]^T -> [H][BT][HS] ---
  gemm_wmma_kernel<EPI_F16><<<dim3(HS_/128, BT_/128, H_), 256, 0, stream>>>(
      ws + o_buf1, 0L, C_, ws + o_wq, (long)HS_*C_, C_,
      ws + o_q, nullptr, (long)BT_*HS_, HS_, nullptr, 0L, nullptr, 0,
      BT_, HS_, C_);
  gemm_wmma_kernel<EPI_BIAS_F16><<<dim3(HS_/128, BT_/128, H_), 256, 0, stream>>>(
      ws + o_buf1, 0L, C_, ws + o_wk, (long)HS_*C_, C_,
      ws + o_k, nullptr, (long)BT_*HS_, HS_, bk, (long)HS_, nullptr, 0,
      BT_, HS_, C_);
  gemm_wmma_kernel<EPI_F16><<<dim3(HS_/128, BT_/128, H_), 256, 0, stream>>>(
      ws + o_buf1, 0L, C_, ws + o_wv, (long)HS_*C_, C_,
      ws + o_v, nullptr, (long)BT_*HS_, HS_, nullptr, 0L, nullptr, 0,
      BT_, HS_, C_);

  // --- attention (scale = C^-0.5 per reference) -> buf1 [BT][C] f16 ---
  attention_kernel<<<dim3(T_/64, B_*H_), 128, 0, stream>>>(
      ws + o_q, ws + o_k, ws + o_v, ws + o_buf1, 0.044194173824159216f);

  // --- x1 = x + attn @ Wproj + bproj -> d_out (f32) ---
  gemm_wmma_kernel<EPI_BIAS_RES_F32><<<dim3(C_/128, BT_/128, 1), 256, 0, stream>>>(
      ws + o_buf1, 0L, C_, ws + o_wp, 0L, C_,
      nullptr, out, 0L, C_, bproj, 0L, x, C_,
      BT_, C_, C_);

  // --- x2 = rmsnorm(x1, w2): f32 in-place into d_out, f16 copy into q-region ---
  rmsnorm_kernel<<<BT_, 256, 0, stream>>>(out, w2, ws + o_q, out);

  // --- dual-branch FFN, chunked over rows; mean folded into Wo scale ---
  const int RB = 32768;                 // rows per chunk (4 chunks)
  for (int r0 = 0; r0 < BT_; r0 += RB) {
    // hmid[RB][NF*FF] = gelu(x2 @ Wi[n]) (branch n lands in columns n*FF..)
    gemm_wmma_kernel<EPI_GELU_F16><<<dim3(FF_/128, RB/128, NF_), 256, 0, stream>>>(
        ws + o_q + (size_t)r0 * C_, 0L, C_, ws + o_wi, (long)FF_*C_, C_,
        ws + o_k, nullptr, (long)FF_, NF_*FF_, nullptr, 0L, nullptr, 0,
        RB, FF_, C_);
    // d_out = x2 + hmid @ Wo_packed   (K = NF*FF = 4096)
    gemm_wmma_kernel<EPI_RES_F32><<<dim3(C_/128, RB/128, 1), 256, 0, stream>>>(
        ws + o_k, 0L, NF_*FF_, ws + o_wo, 0L, NF_*FF_,
        nullptr, out + (size_t)r0 * C_, 0L, C_,
        nullptr, 0L, out + (size_t)r0 * C_, C_,
        RB, C_, NF_*FF_);
  }
}
